// MambaBlock_74208444940358
// MI455X (gfx1250) — matrix operand order repaired and checked by reading, not compiled, not run on hardware
//
#include <hip/hip_runtime.h>
#include <hip/hip_bf16.h>

// ---------------------------------------------------------------------------
// Mamba block for MI455X (gfx1250): bf16 WMMA GEMMs (fp32 accumulate) with
// async global->LDS double-buffered staging + fp32 elementwise/scan kernels.
// ---------------------------------------------------------------------------

typedef __bf16 bf16;
typedef __attribute__((ext_vector_type(16))) __bf16 bf16x16;
typedef __attribute__((ext_vector_type(8)))  __bf16 bf16x8;
typedef __attribute__((ext_vector_type(8)))  float  f32x8;
typedef __attribute__((ext_vector_type(4)))  int    i32x4;
typedef __attribute__((ext_vector_type(4)))  unsigned int u32x4;

#define AS_GLOBAL __attribute__((address_space(1)))
#define AS_LDS    __attribute__((address_space(3)))

#define D_MODEL  1024
#define D_HIDDEN 2048
#define BATCH    4
#define SEQ      4096
#define NTOK     (BATCH * SEQ)          // 16384 tokens

// GEMM tiling: block = 256 threads = 8 wave32 waves; block tile 128x128,
// K-step 32 (one v_wmma_f32_16x16x32_bf16 per 16x16x32 sub-problem).
#define TM   128
#define TN   128
#define TK   32
#define LDST 40   // LDS row stride in bf16 elems (80B) -> conflict-free frags

// ---------------------------------------------------------------------------
// Async global->LDS copy (GLOBAL_LOAD_ASYNC_TO_LDS_B128, ASYNCcnt-tracked).
// Probe-learned prototype: (i32x4 AS1* src, i32x4 AS3* dst, imm off, imm cpol).
// Guarded: falls back to a synchronous 16B copy if the builtin is absent.
// ---------------------------------------------------------------------------
#if defined(__has_builtin)
#if __has_builtin(__builtin_amdgcn_global_load_async_to_lds_b128) && \
    __has_builtin(__builtin_amdgcn_s_wait_asynccnt)
#define USE_ASYNC_LDS 1
#endif
#endif
#ifndef USE_ASYNC_LDS
#define USE_ASYNC_LDS 0
#endif

__device__ __forceinline__ void cp_b128(bf16* lds, const bf16* g) {
#if USE_ASYNC_LDS
  __builtin_amdgcn_global_load_async_to_lds_b128(
      (AS_GLOBAL i32x4*)(void*)g,
      (AS_LDS i32x4*)(void*)lds,
      /*imm offset=*/0, /*cpol=*/0);
#else
  *(u32x4*)lds = *(const u32x4*)g;
#endif
}

__device__ __forceinline__ void wait_async_all() {
#if USE_ASYNC_LDS
  __builtin_amdgcn_s_wait_asynccnt(0);
#endif
}

__device__ __forceinline__ float silu_f(float x) {
  return x / (1.0f + __expf(-x));
}

// ---------------------------------------------------------------------------
// fp32 -> bf16 conversion (weights + activations), grid-stride.
// ---------------------------------------------------------------------------
__global__ void cvt_f32_bf16_kernel(const float* __restrict__ in,
                                    bf16* __restrict__ out, int n) {
  int i = blockIdx.x * blockDim.x + threadIdx.x;
  int stride = gridDim.x * blockDim.x;
  for (; i < n; i += stride) out[i] = (bf16)in[i];
}

// ---------------------------------------------------------------------------
// WMMA fragment loads from LDS, following the CDNA5 16-bit layouts.
// A 16x32 (MxK): lanes 0-15 row M=lane, K=0..7 then K=16..23;
//                lanes 16-31 same rows, K=8..15 then K=24..31.
// B 32x16 (KxN): lane&15 = N; same K order as A (K=0..7,16..23 / 8..15,24..31).
// ---------------------------------------------------------------------------
__device__ __forceinline__ bf16x16 frag_a(const bf16* t, int r0, int lane) {
  const bf16* p = t + (r0 + (lane & 15)) * LDST + ((lane >> 4) << 3);
  bf16x8 lo = *(const bf16x8*)(p);        // K = kb .. kb+7
  bf16x8 hi = *(const bf16x8*)(p + 16);   // K = 16+kb .. 16+kb+7
  return __builtin_shufflevector(lo, hi, 0, 1, 2, 3, 4, 5, 6, 7,
                                 8, 9, 10, 11, 12, 13, 14, 15);
}

__device__ __forceinline__ bf16x16 frag_b(const bf16* t, int r0, int lane) {
  const bf16* p = t + (r0 + (lane & 15)) * LDST + ((lane >> 4) << 3);
  bf16x8 lo = *(const bf16x8*)(p);        // K = kb .. kb+7
  bf16x8 hi = *(const bf16x8*)(p + 16);   // K = 16+kb .. 16+kb+7
  return __builtin_shufflevector(lo, hi, 0, 1, 2, 3, 4, 5, 6, 7,
                                 8, 9, 10, 11, 12, 13, 14, 15);
}

// ---------------------------------------------------------------------------
// Generic GEMM: outF[M,N] (+)= A[M,K] * W[N,K]^T (+ bias), optional bf16
// mirror. A, W bf16 row-major; fp32 WMMA accumulate. Double-buffered LDS
// tiles filled by async global->LDS B128 transfers.
// M % 128 == 0, N % 128 == 0, K % 32 == 0 (true for all shapes here).
// ---------------------------------------------------------------------------
template <bool HAS_BIAS, bool ACC, bool MIRROR>
__global__ __launch_bounds__(256)
void gemm_bf16_wmma_kernel(const bf16* __restrict__ A, const bf16* __restrict__ W,
                           const float* __restrict__ bias,
                           float* __restrict__ outF, bf16* __restrict__ outB,
                           int M, int N, int K) {
  __shared__ bf16 lA[2][TM * LDST];
  __shared__ bf16 lB[2][TN * LDST];

  const int tid   = threadIdx.x;
  const int lane  = tid & 31;
  const int wave  = tid >> 5;
  const int tileM = blockIdx.y * TM;
  const int tileN = blockIdx.x * TN;
  const int wm    = (wave >> 2) * 64;   // 2 waves along M (each 64 rows)
  const int wn    = (wave & 3) * 32;    // 4 waves along N (each 32 cols)

  f32x8 acc[4][2];
#pragma unroll
  for (int i = 0; i < 4; ++i)
#pragma unroll
    for (int j = 0; j < 2; ++j)
#pragma unroll
      for (int e = 0; e < 8; ++e) acc[i][j][e] = 0.0f;

  // Per-tile staging: 512 16-byte chunks (128 rows x 4), 2 per thread.
  auto stage = [&](int buf, int k0) {
#pragma unroll
    for (int c = 0; c < 2; ++c) {
      int chunk = tid + c * 256;
      int row   = chunk >> 2;
      int col   = (chunk & 3) << 3;   // 8 bf16 = 16B per chunk
      cp_b128(&lA[buf][row * LDST + col],
              A + (size_t)(tileM + row) * K + k0 + col);
      cp_b128(&lB[buf][row * LDST + col],
              W + (size_t)(tileN + row) * K + k0 + col);
    }
  };

  stage(0, 0);
  wait_async_all();
  __syncthreads();

  int buf = 0;
  for (int k0 = 0; k0 < K; k0 += TK) {
    const bool more = (k0 + TK) < K;
    if (more) stage(buf ^ 1, k0 + TK);
    if (k0 + 2 * TK < K) {  // pull the K-step after next into GL2
      __builtin_prefetch(A + (size_t)(tileM + (tid >> 1)) * K + k0 + 2 * TK, 0, 1);
      __builtin_prefetch(W + (size_t)(tileN + (tid >> 1)) * K + k0 + 2 * TK, 0, 1);
    }

    bf16x16 af[4], bfr[2];
#pragma unroll
    for (int i = 0; i < 4; ++i) af[i] = frag_a(lA[buf], wm + i * 16, lane);
#pragma unroll
    for (int j = 0; j < 2; ++j) bfr[j] = frag_b(lB[buf], wn + j * 16, lane);

#pragma unroll
    for (int i = 0; i < 4; ++i)
#pragma unroll
      for (int j = 0; j < 2; ++j)
        acc[i][j] = __builtin_amdgcn_wmma_f32_16x16x32_bf16(
            /*neg_a=*/false, af[i], /*neg_b=*/false, bfr[j],
            /*c_mod=*/(short)0, acc[i][j],
            /*reuse_a=*/false, /*reuse_b=*/false);

    wait_async_all();
    __syncthreads();
    buf ^= 1;
  }

  // Epilogue. D layout: VGPR v, lanes 0-15 -> (M=v, N=lane),
  // lanes 16-31 -> (M=8+v, N=lane-16).
  const int nlo = lane & 15;
  const int mhi = (lane >> 4) << 3;
#pragma unroll
  for (int i = 0; i < 4; ++i) {
#pragma unroll
    for (int j = 0; j < 2; ++j) {
      const int col  = tileN + wn + j * 16 + nlo;
      const int rowb = tileM + wm + i * 16 + mhi;
      const float bv = HAS_BIAS ? bias[col] : 0.0f;
#pragma unroll
      for (int v = 0; v < 8; ++v) {
        size_t o = (size_t)(rowb + v) * N + col;
        float r = acc[i][j][v] + bv;
        if constexpr (ACC) r += outF[o];
        outF[o] = r;
        if constexpr (MIRROR) outB[o] = (bf16)r;
      }
    }
  }
}

// ---------------------------------------------------------------------------
// Depthwise conv1d (k=3, pad=1, per-channel) over L, then SiLU -> bf16.
// y layout [B, L, D]; neighbors along L are +/- D_MODEL apart.
// ---------------------------------------------------------------------------
__global__ void conv_silu_kernel(const float* __restrict__ y,
                                 const float* __restrict__ cw,   // [D,1,3]
                                 const float* __restrict__ cb,   // [D]
                                 bf16* __restrict__ out) {
  const int n = NTOK * D_MODEL;
  int i = blockIdx.x * blockDim.x + threadIdx.x;
  int stride = gridDim.x * blockDim.x;
  for (; i < n; i += stride) {
    int d = i % D_MODEL;
    int l = (i / D_MODEL) % SEQ;
    float left  = (l > 0)       ? y[i - D_MODEL] : 0.0f;
    float right = (l < SEQ - 1) ? y[i + D_MODEL] : 0.0f;
    float v = cw[d * 3 + 0] * left + cw[d * 3 + 1] * y[i] +
              cw[d * 3 + 2] * right + cb[d];
    out[i] = (bf16)silu_f(v);
  }
}

// ---------------------------------------------------------------------------
// Fused SSM scan: per (b,h) channel, sequential over L (8192 independent
// scans; consecutive h -> consecutive addresses -> coalesced).
// ---------------------------------------------------------------------------
__global__ void ssm_scan_kernel(const float* __restrict__ dl,
                                const float* __restrict__ At,
                                const float* __restrict__ Bx,
                                bf16* __restrict__ hb) {
  int id = blockIdx.x * blockDim.x + threadIdx.x;
  if (id >= BATCH * D_HIDDEN) return;
  int b = id / D_HIDDEN;
  int h = id % D_HIDDEN;
  size_t base = (size_t)b * SEQ * D_HIDDEN + h;
  float p = 1.0f, s = 0.0f;
  for (int l = 0; l < SEQ; ++l) {
    size_t ix = base + (size_t)l * D_HIDDEN;
    float dv = dl[ix];
    float sp = (dv > 20.0f) ? dv : log1pf(__expf(dv));   // softplus
    float a  = __expf(-sp * At[ix]);
    p *= a;
    s += Bx[ix] * p;
    hb[ix] = (bf16)s;
  }
}

// ---------------------------------------------------------------------------
// Gate: z = ssm * silu(t) -> bf16
// ---------------------------------------------------------------------------
__global__ void gate_kernel(const float* __restrict__ ssm,
                            const float* __restrict__ t,
                            bf16* __restrict__ zb, int n) {
  int i = blockIdx.x * blockDim.x + threadIdx.x;
  int stride = gridDim.x * blockDim.x;
  for (; i < n; i += stride) zb[i] = (bf16)(ssm[i] * silu_f(t[i]));
}

// ---------------------------------------------------------------------------
// Host orchestration. Workspace layout (~700 MB assumed available in d_ws).
// ---------------------------------------------------------------------------
extern "C" void kernel_launch(void* const* d_in, const int* in_sizes, int n_in,
                              void* d_out, int out_size, void* d_ws, size_t ws_size,
                              hipStream_t stream) {
  (void)in_sizes; (void)n_in; (void)out_size; (void)ws_size;

  const float* x      = (const float*)d_in[0];
  const float* w1     = (const float*)d_in[1];
  const float* w2     = (const float*)d_in[2];
  const float* w3     = (const float*)d_in[3];
  const float* conv_w = (const float*)d_in[4];
  const float* conv_b = (const float*)d_in[5];
  const float* A_w    = (const float*)d_in[6];
  const float* A_b    = (const float*)d_in[7];
  const float* B_w    = (const float*)d_in[8];
  const float* B_b    = (const float*)d_in[9];
  const float* C_w    = (const float*)d_in[10];
  const float* C_b    = (const float*)d_in[11];
  const float* D_w    = (const float*)d_in[12];
  const float* D_b    = (const float*)d_in[13];
  const float* dl_w   = (const float*)d_in[14];
  const float* dl_b   = (const float*)d_in[15];

  float* out = (float*)d_out;   // also serves as fp32 ssm buffer

  char* ws = (char*)d_ws;
  size_t off = 0;
  auto alloc = [&](size_t bytes) -> void* {
    void* p = ws + off;
    off += (bytes + 255) & ~(size_t)255;
    return p;
  };

  const int nMD  = NTOK * D_MODEL;    // 16,777,216
  const int nMH  = NTOK * D_HIDDEN;   // 33,554,432
  const int nDD  = D_MODEL * D_MODEL;
  const int nHD  = D_HIDDEN * D_MODEL;

  bf16* xb0   = (bf16*)alloc(sizeof(bf16) * nMD);   // bf16(x)
  bf16* xb    = (bf16*)alloc(sizeof(bf16) * nMD);   // bf16(silu(conv(x@w1)))
  bf16* hb    = (bf16*)alloc(sizeof(bf16) * nMH);   // bf16(h)
  bf16* ssmb  = (bf16*)alloc(sizeof(bf16) * nMD);   // bf16(ssm_out)
  bf16* zb    = (bf16*)alloc(sizeof(bf16) * nMD);   // bf16(z)
  bf16* w1b   = (bf16*)alloc(sizeof(bf16) * nDD);
  bf16* w2b   = (bf16*)alloc(sizeof(bf16) * nDD);
  bf16* w3b   = (bf16*)alloc(sizeof(bf16) * nDD);
  bf16* Awb   = (bf16*)alloc(sizeof(bf16) * nHD);
  bf16* Bwb   = (bf16*)alloc(sizeof(bf16) * nHD);
  bf16* Cwb   = (bf16*)alloc(sizeof(bf16) * nHD);   // [D_MODEL, D_HIDDEN]
  bf16* Dwb   = (bf16*)alloc(sizeof(bf16) * nDD);
  bf16* dlwb  = (bf16*)alloc(sizeof(bf16) * nHD);
  float* y1   = (float*)alloc(sizeof(float) * nMD); // x@w1 (reused as t)
  float* dl   = (float*)alloc(sizeof(float) * nMH);
  float* At   = (float*)alloc(sizeof(float) * nMH);
  float* Bx   = (float*)alloc(sizeof(float) * nMH);

  auto cvt = [&](const float* src, bf16* dst, int n) {
    int blocks = (n + 256 * 16 - 1) / (256 * 16);
    cvt_f32_bf16_kernel<<<dim3(blocks), dim3(256), 0, stream>>>(src, dst, n);
  };
  auto grd = [](int M, int N) { return dim3(N / TN, M / TM); };
  const dim3 blk(256);

  // --- convert inputs to bf16 ---
  cvt(x, xb0, nMD);
  cvt(w1, w1b, nDD);
  cvt(w2, w2b, nDD);
  cvt(w3, w3b, nDD);
  cvt(A_w, Awb, nHD);
  cvt(B_w, Bwb, nHD);
  cvt(C_w, Cwb, nHD);
  cvt(D_w, Dwb, nDD);
  cvt(dl_w, dlwb, nHD);

  // --- x = x @ w1^T ---
  gemm_bf16_wmma_kernel<false, false, false>
      <<<grd(NTOK, D_MODEL), blk, 0, stream>>>(xb0, w1b, nullptr, y1, nullptr,
                                               NTOK, D_MODEL, D_MODEL);

  // --- depthwise conv + silu -> xb (bf16) ---
  conv_silu_kernel<<<dim3(8192), blk, 0, stream>>>(y1, conv_w, conv_b, xb);

  // --- the three H-dim projections ---
  gemm_bf16_wmma_kernel<true, false, false>
      <<<grd(NTOK, D_HIDDEN), blk, 0, stream>>>(xb, dlwb, dl_b, dl, nullptr,
                                                NTOK, D_HIDDEN, D_MODEL);
  gemm_bf16_wmma_kernel<true, false, false>
      <<<grd(NTOK, D_HIDDEN), blk, 0, stream>>>(xb, Awb, A_b, At, nullptr,
                                                NTOK, D_HIDDEN, D_MODEL);
  gemm_bf16_wmma_kernel<true, false, false>
      <<<grd(NTOK, D_HIDDEN), blk, 0, stream>>>(xb, Bwb, B_b, Bx, nullptr,
                                                NTOK, D_HIDDEN, D_MODEL);

  // --- fused softplus/exp/cumprod/cumsum scan -> hb (bf16) ---
  ssm_scan_kernel<<<dim3((BATCH * D_HIDDEN) / 256), blk, 0, stream>>>(dl, At,
                                                                      Bx, hb);

  // --- ssm_out = h @ C^T + C_b + x @ D^T + D_b  (fp32 in d_out, bf16 mirror) ---
  gemm_bf16_wmma_kernel<true, false, false>
      <<<grd(NTOK, D_MODEL), blk, 0, stream>>>(hb, Cwb, C_b, out, nullptr,
                                               NTOK, D_MODEL, D_HIDDEN);
  gemm_bf16_wmma_kernel<true, true, true>
      <<<grd(NTOK, D_MODEL), blk, 0, stream>>>(xb, Dwb, D_b, out, ssmb,
                                               NTOK, D_MODEL, D_MODEL);

  // --- gated output: t = ssm@w2^T ; z = ssm*silu(t) ; out += z@w3^T ---
  gemm_bf16_wmma_kernel<false, false, false>
      <<<grd(NTOK, D_MODEL), blk, 0, stream>>>(ssmb, w2b, nullptr, y1, nullptr,
                                               NTOK, D_MODEL, D_MODEL);
  gate_kernel<<<dim3(4096), blk, 0, stream>>>(out, y1, zb, nMD);
  gemm_bf16_wmma_kernel<false, true, false>
      <<<grd(NTOK, D_MODEL), blk, 0, stream>>>(zb, w3b, nullptr, out, nullptr,
                                               NTOK, D_MODEL, D_MODEL);
}
